// CausalSelfAttention_8057358647352
// MI455X (gfx1250) — compile-verified, matrix operand order checked by reading, not run on hardware
//
#include <hip/hip_runtime.h>
#include <hip/hip_bf16.h>

typedef _Float16 h16;
typedef __attribute__((ext_vector_type(4)))  _Float16 v4h;
typedef __attribute__((ext_vector_type(16))) _Float16 v16h;
typedef __attribute__((ext_vector_type(8)))  float    v8f;

#define WMMA(a, b, c) __builtin_amdgcn_wmma_f32_16x16x32_f16(false, (a), false, (b), (short)0, (c), false, false)

static constexpr int Bz = 2, Tt = 2048, Cc = 1024, Hh = 16, Dh = 64;
static constexpr int Mrows = Bz * Tt;  // 4096

__device__ __forceinline__ v8f vzero() {
    v8f z = {0.f, 0.f, 0.f, 0.f, 0.f, 0.f, 0.f, 0.f};
    return z;
}

// ---------------- f32 -> f16 convert (4 elems/thread) ----------------
__global__ void cvt_kernel(const float* __restrict__ src, h16* __restrict__ dst, int n4) {
    int i = blockIdx.x * blockDim.x + threadIdx.x;
    if (i < n4) {
        float4 v = reinterpret_cast<const float4*>(src)[i];
        v4h o = {(h16)v.x, (h16)v.y, (h16)v.z, (h16)v.w};
        reinterpret_cast<v4h*>(dst)[i] = o;
    }
}

// ---------------- fused QKV projection: y = x @ W^T ----------------
// All 4 waves of a block share the same 16-row A tile -> stage it once in LDS
// via CDNA5 async global->LDS copy (ASYNCcnt), then stream B from global.
__global__ void __launch_bounds__(128) qkv_gemm(const h16* __restrict__ X,
                                                const h16* __restrict__ Wq,
                                                const h16* __restrict__ Wk,
                                                const h16* __restrict__ Wv,
                                                h16* __restrict__ Qo,
                                                h16* __restrict__ Ko,
                                                h16* __restrict__ Vto) {
    __shared__ __align__(16) h16 atile[16 * Cc];  // 32 KB: 16 rows x 1024
    const int wv = threadIdx.x >> 5;
    const int lane = threadIdx.x & 31;
    const int r = lane & 15, g = lane >> 4;
    const int wave = blockIdx.x * 4 + wv;
    const int mt = wave >> 4;   // same for all 4 waves in block
    const int nb = wave & 15;   // col block (== head index)
    const h16* Ws[3] = {Wq, Wk, Wv};

    // cooperative async stage of A tile: 32 KB = 128 threads x 16 x b128
    {
        const h16* gbase = X + (size_t)mt * 16 * Cc;
#pragma unroll
        for (int it = 0; it < 16; ++it) {
            int idx = (it * 128 + threadIdx.x) * 8;  // h16 elements, 16B chunks
            unsigned la = (unsigned)(uintptr_t)(atile + idx);
            const h16* ga = gbase + idx;
            asm volatile("global_load_async_to_lds_b128 %0, %1, off" :: "v"(la), "v"(ga) : "memory");
        }
        asm volatile("s_wait_asynccnt 0x0" ::: "memory");
        __syncthreads();
    }

    v8f acc[3][4];
#pragma unroll
    for (int w = 0; w < 3; w++)
#pragma unroll
        for (int nt = 0; nt < 4; nt++) acc[w][nt] = vzero();

    const h16* alds = atile + r * Cc + g * 16;
    for (int k0 = 0; k0 < Cc; k0 += 32) {
        v16h a = *(const v16h*)(alds + k0);
#pragma unroll
        for (int w = 0; w < 3; w++) {
            const h16* wb = Ws[w] + (nb * 64 + r) * Cc + g * 16 + k0;
#pragma unroll
            for (int nt = 0; nt < 4; nt++) {
                v16h bf = *(const v16h*)(wb + nt * 16 * Cc);
                acc[w][nt] = WMMA(a, bf, acc[w][nt]);
            }
        }
    }
#pragma unroll
    for (int nt = 0; nt < 4; nt++) {
        int d = nt * 16 + r;
#pragma unroll
        for (int j = 0; j < 8; j++) {
            int m = mt * 16 + j + 8 * g;
            int b = m >> 11, t = m & (Tt - 1);
            int qk = ((b * Hh + nb) * Tt + t) * Dh + d;
            Qo[qk] = (h16)acc[0][nt][j];
            Ko[qk] = (h16)acc[1][nt][j];
            Vto[((b * Hh + nb) * Dh + d) * Tt + t] = (h16)acc[2][nt][j];  // V transposed
        }
    }
}

// ---------------- flash attention: one wave per (b,h,16 query rows) ----------------
// 64-key chunks: 16 WMMAs per chunk, softmax reduction cost amortized 2x vs 32.
__global__ void __launch_bounds__(128) attn_kernel(const h16* __restrict__ Q,
                                                   const h16* __restrict__ K,
                                                   const h16* __restrict__ Vt,
                                                   h16* __restrict__ O) {
    __shared__ __align__(64) h16 pst[4][16][64];  // per-wave P staging, 8 KB total
    const int wv = threadIdx.x >> 5;
    const int lane = threadIdx.x & 31;
    const int r = lane & 15, g = lane >> 4;
    const int wave = blockIdx.x * 4 + wv;
    const int qb = wave & 127;  // T/16 query blocks
    const int bh = wave >> 7;   // 0..31
    const int q0 = qb * 16;

    const h16* Qb = Q + bh * Tt * Dh;
    const h16* Kb = K + bh * Tt * Dh;
    const h16* Vb = Vt + bh * Dh * Tt;

    v16h qa0 = *(const v16h*)(Qb + (q0 + r) * Dh + g * 16);
    v16h qa1 = *(const v16h*)(Qb + (q0 + r) * Dh + 32 + g * 16);

    float mi[8], li[8];
    v8f oacc[4];
#pragma unroll
    for (int j = 0; j < 8; j++) { mi[j] = -1e30f; li[j] = 0.f; }
#pragma unroll
    for (int dt = 0; dt < 4; dt++) oacc[dt] = vzero();

    const float LOG2E = 1.4426950408889634f;

    for (int key0 = 0; key0 < q0 + 16; key0 += 64) {
        // S chunk = 16 queries x 64 keys: 4 tiles, K=64 over d
        v8f s[4];
#pragma unroll
        for (int nt = 0; nt < 4; nt++) {
            s[nt] = vzero();
            const h16* kb = Kb + (key0 + nt * 16 + r) * Dh + g * 16;
            v16h b0 = *(const v16h*)(kb);
            v16h b1 = *(const v16h*)(kb + 32);
            s[nt] = WMMA(qa0, b0, s[nt]);
            s[nt] = WMMA(qa1, b1, s[nt]);
        }
#pragma unroll
        for (int nt = 0; nt < 4; nt++)
#pragma unroll
            for (int j = 0; j < 8; j++) s[nt][j] *= 0.125f;  // 1/sqrt(64)

        if (key0 + 63 > q0) {  // causal mask only on diagonal chunks
#pragma unroll
            for (int nt = 0; nt < 4; nt++)
#pragma unroll
                for (int j = 0; j < 8; j++) {
                    int qrow = q0 + j + 8 * g;
                    if (key0 + nt * 16 + r > qrow) s[nt][j] = -1e30f;
                }
        }

        float al[8];
#pragma unroll
        for (int j = 0; j < 8; j++) {
            float cm = fmaxf(fmaxf(s[0][j], s[1][j]), fmaxf(s[2][j], s[3][j]));
            cm = fmaxf(cm, __shfl_xor(cm, 1, 32));
            cm = fmaxf(cm, __shfl_xor(cm, 2, 32));
            cm = fmaxf(cm, __shfl_xor(cm, 4, 32));
            cm = fmaxf(cm, __shfl_xor(cm, 8, 32));
            float m2 = fmaxf(mi[j], cm);
            al[j] = __builtin_exp2f((mi[j] - m2) * LOG2E);
            s[0][j] = __builtin_exp2f((s[0][j] - m2) * LOG2E);
            s[1][j] = __builtin_exp2f((s[1][j] - m2) * LOG2E);
            s[2][j] = __builtin_exp2f((s[2][j] - m2) * LOG2E);
            s[3][j] = __builtin_exp2f((s[3][j] - m2) * LOG2E);
            float rs = (s[0][j] + s[1][j]) + (s[2][j] + s[3][j]);
            rs += __shfl_xor(rs, 1, 32);
            rs += __shfl_xor(rs, 2, 32);
            rs += __shfl_xor(rs, 4, 32);
            rs += __shfl_xor(rs, 8, 32);
            li[j] = li[j] * al[j] + rs;
            mi[j] = m2;
        }
#pragma unroll
        for (int dt = 0; dt < 4; dt++)
#pragma unroll
            for (int j = 0; j < 8; j++) oacc[dt][j] *= al[j];

        // P: C-layout -> row-major LDS -> two A-layout fragments (K=64)
#pragma unroll
        for (int nt = 0; nt < 4; nt++)
#pragma unroll
            for (int j = 0; j < 8; j++) pst[wv][j + 8 * g][nt * 16 + r] = (h16)s[nt][j];
        asm volatile("s_wait_dscnt 0" ::: "memory");
        v16h pa0 = *(const v16h*)(&pst[wv][r][g * 16]);
        v16h pa1 = *(const v16h*)(&pst[wv][r][32 + g * 16]);
#pragma unroll
        for (int dt = 0; dt < 4; dt++) {
            const h16* vrow = Vb + (dt * 16 + r) * Tt + key0 + g * 16;
            v16h vb0 = *(const v16h*)(vrow);
            v16h vb1 = *(const v16h*)(vrow + 32);
            oacc[dt] = WMMA(pa0, vb0, oacc[dt]);
            oacc[dt] = WMMA(pa1, vb1, oacc[dt]);
        }
    }

    // epilogue: write O as [B,T,H,D] (== row-major [4096,1024]) in f16
    int b = bh >> 4, h = bh & 15;
#pragma unroll
    for (int j = 0; j < 8; j++) {
        float inv = 1.0f / li[j];
        int t = q0 + j + 8 * g;
#pragma unroll
        for (int dt = 0; dt < 4; dt++) {
            O[((b * Tt + t) * Hh + h) * Dh + dt * 16 + r] = (h16)(oacc[dt][j] * inv);
        }
    }
}

// ---------------- output projection: out = A @ Wp^T + bp (f32 out) ----------------
__global__ void __launch_bounds__(128) proj_gemm(const h16* __restrict__ A,
                                                 const h16* __restrict__ W,
                                                 const float* __restrict__ bias,
                                                 float* __restrict__ out) {
    const int wave = blockIdx.x * (blockDim.x >> 5) + (threadIdx.x >> 5);
    const int lane = threadIdx.x & 31;
    const int r = lane & 15, g = lane >> 4;
    const int mt = wave >> 4;
    const int nb = wave & 15;

    v8f acc[4];
#pragma unroll
    for (int nt = 0; nt < 4; nt++) acc[nt] = vzero();

    const h16* arow = A + (mt * 16 + r) * Cc + g * 16;
    const h16* wrow = W + (nb * 64 + r) * Cc + g * 16;
    for (int k0 = 0; k0 < Cc; k0 += 32) {
        v16h a = *(const v16h*)(arow + k0);
#pragma unroll
        for (int nt = 0; nt < 4; nt++) {
            v16h bf = *(const v16h*)(wrow + nt * 16 * Cc + k0);
            acc[nt] = WMMA(a, bf, acc[nt]);
        }
    }
#pragma unroll
    for (int nt = 0; nt < 4; nt++) {
        int n = nb * 64 + nt * 16 + r;
        float bb = bias[n];
#pragma unroll
        for (int j = 0; j < 8; j++) {
            int m = mt * 16 + j + 8 * g;
            out[m * Cc + n] = acc[nt][j] + bb;
        }
    }
}

extern "C" void kernel_launch(void* const* d_in, const int* in_sizes, int n_in,
                              void* d_out, int out_size, void* d_ws, size_t ws_size,
                              hipStream_t stream) {
    const float* x = (const float*)d_in[0];
    const float* Wq = (const float*)d_in[1];
    const float* Wk = (const float*)d_in[2];
    const float* Wv = (const float*)d_in[3];
    const float* Wp = (const float*)d_in[4];
    const float* bp = (const float*)d_in[5];
    float* out = (float*)d_out;

    char* ws = (char*)d_ws;
    h16* xh = (h16*)ws;   ws += (size_t)Mrows * Cc * sizeof(h16);   // 8 MB
    h16* wqh = (h16*)ws;  ws += (size_t)Cc * Cc * sizeof(h16);      // 2 MB
    h16* wkh = (h16*)ws;  ws += (size_t)Cc * Cc * sizeof(h16);
    h16* wvh = (h16*)ws;  ws += (size_t)Cc * Cc * sizeof(h16);
    h16* wph = (h16*)ws;  ws += (size_t)Cc * Cc * sizeof(h16);
    h16* Qh = (h16*)ws;   ws += (size_t)Mrows * Cc * sizeof(h16);   // [B,H,T,D]
    h16* Kh = (h16*)ws;   ws += (size_t)Mrows * Cc * sizeof(h16);   // [B,H,T,D]
    h16* Vth = (h16*)ws;  ws += (size_t)Mrows * Cc * sizeof(h16);   // [B,H,D,T]
    h16* Oh = (h16*)ws;   ws += (size_t)Mrows * Cc * sizeof(h16);   // [B,T,H,D]

    int nx4 = Mrows * Cc / 4;
    int nw4 = Cc * Cc / 4;
    cvt_kernel<<<(nx4 + 255) / 256, 256, 0, stream>>>(x, xh, nx4);
    cvt_kernel<<<(nw4 + 255) / 256, 256, 0, stream>>>(Wq, wqh, nw4);
    cvt_kernel<<<(nw4 + 255) / 256, 256, 0, stream>>>(Wk, wkh, nw4);
    cvt_kernel<<<(nw4 + 255) / 256, 256, 0, stream>>>(Wv, wvh, nw4);
    cvt_kernel<<<(nw4 + 255) / 256, 256, 0, stream>>>(Wp, wph, nw4);

    // 4096 waves each, 4 waves per 128-thread block
    qkv_gemm<<<1024, 128, 0, stream>>>(xh, wqh, wkh, wvh, Qh, Kh, Vth);
    attn_kernel<<<1024, 128, 0, stream>>>(Qh, Kh, Vth, Oh);
    proj_gemm<<<1024, 128, 0, stream>>>(Oh, wph, bp, out);
}
